// B3DBModel_40973988004672
// MI455X (gfx1250) — matrix operand order repaired and checked by reading, not compiled, not run on hardware
//
#include <hip/hip_runtime.h>

#define NNODES 20000
#define NEDGES 320000
#define NGRAPH 128
#define DIN    128
#define HDIM   128
#define HEADS  8
#define OUTD   2

// ---------------- WMMA types ----------------
typedef __attribute__((ext_vector_type(16))) __bf16       v16bf;
typedef __attribute__((ext_vector_type(8)))  float        v8f;
typedef __attribute__((ext_vector_type(4)))  unsigned int v4u;

union Frag { v16bf v; v4u q[2]; };

__device__ __forceinline__ unsigned short f32_to_bf16(float f) {
  unsigned u = __float_as_uint(f);
  u += 0x7fffu + ((u >> 16) & 1u);        // round-to-nearest-even
  return (unsigned short)(u >> 16);
}
// monotone float<->uint map for atomicMax on floats
__device__ __forceinline__ unsigned f2o(float f) {
  unsigned u = __float_as_uint(f);
  return (u & 0x80000000u) ? ~u : (u | 0x80000000u);
}
__device__ __forceinline__ float o2f(unsigned u) {
  unsigned v = (u & 0x80000000u) ? (u & 0x7fffffffu) : ~u;
  return __uint_as_float(v);
}

// ---------------- generic elementwise ----------------
__global__ void k_copy_f32(float* __restrict__ o, const float* __restrict__ i, size_t n) {
  size_t t = (size_t)blockIdx.x * blockDim.x + threadIdx.x;
  if (t < n) o[t] = i[t];
}
__global__ void k_fill_f32(float* __restrict__ o, float v, size_t n) {
  size_t t = (size_t)blockIdx.x * blockDim.x + threadIdx.x;
  if (t < n) o[t] = v;
}
__global__ void k_fill_u32(unsigned* __restrict__ o, unsigned v, size_t n) {
  size_t t = (size_t)blockIdx.x * blockDim.x + threadIdx.x;
  if (t < n) o[t] = v;
}
__global__ void k_cvt_bf16(const float* __restrict__ i, unsigned short* __restrict__ o, size_t n) {
  size_t t = (size_t)blockIdx.x * blockDim.x + threadIdx.x;
  if (t < n) o[t] = f32_to_bf16(i[t]);
}
// W[K x Nn] (row-major) -> Wt[Nn x K] bf16 (row-major), so B-columns are contiguous
__global__ void k_cvt_transpose_bf16(const float* __restrict__ W, unsigned short* __restrict__ Wt,
                                     int K, int Nn) {
  size_t t = (size_t)blockIdx.x * blockDim.x + threadIdx.x;
  size_t n = (size_t)K * Nn;
  if (t >= n) return;
  int k = (int)(t / Nn), c = (int)(t % Nn);
  Wt[(size_t)c * K + k] = f32_to_bf16(W[t]);
}

// ---------------- WMMA GEMM: C = A(MxK) * B(KxNn) [+bias][relu] ----------------
// A: bf16 row-major; Bt: bf16 B-transposed (Nn x K row-major). Output f32 and/or bf16.
__global__ void __launch_bounds__(256)
k_gemm_bf16_wmma(const unsigned short* __restrict__ A,
                 const unsigned short* __restrict__ Bt,
                 const float* __restrict__ bias,
                 float* __restrict__ Cf, unsigned short* __restrict__ Cb,
                 int M, int K, int Nn, int relu) {
  int wid  = (int)((blockIdx.x * blockDim.x + threadIdx.x) >> 5);
  int lane = threadIdx.x & 31;
  int tiles_n = Nn >> 4;
  int tiles_m = (M + 15) >> 4;
  if (wid >= tiles_m * tiles_n) return;
  int tm = wid / tiles_n, tn = wid % tiles_n;
  int r = lane & 15, g = lane >> 4;

  int arow = tm * 16 + r; if (arow >= M) arow = M - 1;
  const unsigned short* Ap = A  + (size_t)arow * K;
  const unsigned short* Bp = Bt + (size_t)(tn * 16 + r) * K;

  v8f acc = {};
  for (int k0 = 0; k0 < K; k0 += 32) {
    Frag a, b;
    // ISA 16-bit A 16x32 layout: lane-half g, runs [k0+8g .. +7] and [k0+16+8g .. +7]
    a.q[0] = *reinterpret_cast<const v4u*>(Ap + k0 + g * 8);
    a.q[1] = *reinterpret_cast<const v4u*>(Ap + k0 + 16 + g * 8);
    // B 32x16 layout: lane-half g, runs [k0+8g .. +7] and [k0+16+8g .. +7] of column n (matches A)
    b.q[0] = *reinterpret_cast<const v4u*>(Bp + k0 + g * 8);
    b.q[1] = *reinterpret_cast<const v4u*>(Bp + k0 + 16 + g * 8);
    acc = __builtin_amdgcn_wmma_f32_16x16x32_bf16(false, a.v, false, b.v,
                                                  (short)0, acc, false, false);
  }

  int col = tn * 16 + r;
  float bv = bias ? bias[col] : 0.0f;
  #pragma unroll
  for (int v = 0; v < 8; ++v) {
    int row = tm * 16 + v + 8 * g;   // C/D f32 layout: VGPR v, lane-half g -> M = v + 8g
    if (row >= M) continue;
    float val = acc[v] + bv;
    if (relu && val < 0.0f) val = 0.0f;
    size_t off = (size_t)row * Nn + col;
    if (Cf) Cf[off] = val;
    if (Cb) Cb[off] = f32_to_bf16(val);
  }
}

// ---------------- GIN aggregation: z[dst] += h[src] ----------------
__global__ void k_gin_scatter(const float* __restrict__ h, const int* __restrict__ src,
                              const int* __restrict__ dst, float* __restrict__ z,
                              int E, int D) {
  size_t t = (size_t)blockIdx.x * blockDim.x + threadIdx.x;
  if (t >= (size_t)E * D) return;
  int e = (int)(t / D), c = (int)(t % D);
  atomicAdd(&z[(size_t)dst[e] * D + c], h[(size_t)src[e] * D + c]);
}

// ---------------- BatchNorm (batch statistics) ----------------
__global__ void k_bn_stats(const float* __restrict__ h, float* __restrict__ mean,
                           float* __restrict__ rstd, int Nr, int D) {
  __shared__ float s1[256], s2[256];
  int d = blockIdx.x, t = threadIdx.x;
  float s = 0.f, ss = 0.f;
  for (int n = t; n < Nr; n += 256) {
    float v = h[(size_t)n * D + d];
    s += v; ss += v * v;
  }
  s1[t] = s; s2[t] = ss; __syncthreads();
  for (int o = 128; o > 0; o >>= 1) {
    if (t < o) { s1[t] += s1[t + o]; s2[t] += s2[t + o]; }
    __syncthreads();
  }
  if (t == 0) {
    float mu = s1[0] / Nr;
    float var = s2[0] / Nr - mu * mu;
    if (var < 0.f) var = 0.f;
    mean[d] = mu;
    rstd[d] = rsqrtf(var + 1e-5f);
  }
}
__global__ void k_bn_apply_relu(float* __restrict__ h, const float* __restrict__ mean,
                                const float* __restrict__ rstd, const float* __restrict__ gm,
                                const float* __restrict__ bt, int Nr, int D) {
  size_t t = (size_t)blockIdx.x * blockDim.x + threadIdx.x;
  if (t >= (size_t)Nr * D) return;
  int d = (int)(t % D);
  float v = gm[d] * (h[t] - mean[d]) * rstd[d] + bt[d];
  h[t] = v > 0.f ? v : 0.f;
}

// ---------------- GAT ----------------
__global__ void k_gat_coef(const float* __restrict__ xh, const float* __restrict__ as,
                           const float* __restrict__ ad, float* __restrict__ ao,
                           float* __restrict__ bo) {
  int t = blockIdx.x * blockDim.x + threadIdx.x;
  if (t >= NNODES * HEADS) return;
  int n = t >> 3, hd = t & 7;
  const float* p = xh + (size_t)n * (HEADS * HDIM) + hd * HDIM;
  const float* a1 = as + hd * HDIM;
  const float* a2 = ad + hd * HDIM;
  float s1 = 0.f, s2 = 0.f;
  for (int c = 0; c < HDIM; ++c) { s1 += p[c] * a1[c]; s2 += p[c] * a2[c]; }
  ao[t] = s1; bo[t] = s2;
}
__global__ void k_gat_logits(const float* __restrict__ ao, const float* __restrict__ bo,
                             const int* __restrict__ src, const int* __restrict__ dst,
                             float* __restrict__ elog, unsigned* __restrict__ smax) {
  size_t t = (size_t)blockIdx.x * blockDim.x + threadIdx.x;
  if (t >= (size_t)(NEDGES + NNODES) * HEADS) return;
  int ei = (int)(t >> 3), hd = (int)(t & 7);
  int s = ei < NEDGES ? src[ei] : ei - NEDGES;
  int d = ei < NEDGES ? dst[ei] : ei - NEDGES;
  float v = ao[s * HEADS + hd] + bo[d * HEADS + hd];
  v = v > 0.f ? v : 0.2f * v;                     // leaky_relu(0.2)
  elog[t] = v;
  atomicMax(&smax[d * HEADS + hd], f2o(v));
}
__global__ void k_gat_expsum(float* __restrict__ elog, const int* __restrict__ dst,
                             const unsigned* __restrict__ smax, float* __restrict__ ssum) {
  size_t t = (size_t)blockIdx.x * blockDim.x + threadIdx.x;
  if (t >= (size_t)(NEDGES + NNODES) * HEADS) return;
  int ei = (int)(t >> 3), hd = (int)(t & 7);
  int d = ei < NEDGES ? dst[ei] : ei - NEDGES;
  float ex = __expf(elog[t] - o2f(smax[d * HEADS + hd]));
  elog[t] = ex;
  atomicAdd(&ssum[d * HEADS + hd], ex);
}
__global__ void k_gat_aggregate(const float* __restrict__ elog, const float* __restrict__ ssum,
                                const float* __restrict__ xh, const int* __restrict__ src,
                                const int* __restrict__ dst, float* __restrict__ outh) {
  size_t t = (size_t)blockIdx.x * blockDim.x + threadIdx.x;
  if (t >= (size_t)(NEDGES + NNODES) * HDIM) return;
  int ei = (int)(t >> 7), c = (int)(t & 127);
  int s = ei < NEDGES ? src[ei] : ei - NEDGES;
  int d = ei < NEDGES ? dst[ei] : ei - NEDGES;
  float acc = 0.f;
  #pragma unroll
  for (int hd = 0; hd < HEADS; ++hd) {
    float alpha = elog[(size_t)ei * HEADS + hd] / (ssum[d * HEADS + hd] + 1e-16f);
    acc += alpha * xh[(size_t)s * (HEADS * HDIM) + hd * HDIM + c];
  }
  atomicAdd(&outh[(size_t)d * HDIM + c], acc);
}
__global__ void k_gat_finalize(float* __restrict__ h, const float* __restrict__ bias) {
  size_t t = (size_t)blockIdx.x * blockDim.x + threadIdx.x;
  if (t >= (size_t)NNODES * HDIM) return;
  h[t] = h[t] * (1.0f / HEADS) + bias[t & 127];   // mean over heads + gat_bias
}

// ---------------- Global attention pooling ----------------
__global__ void k_gate_logits(const float* __restrict__ h, const float* __restrict__ gw,
                              const float* __restrict__ gb, const int* __restrict__ batch,
                              float* __restrict__ gate, unsigned* __restrict__ gmax) {
  int n = blockIdx.x * blockDim.x + threadIdx.x;
  if (n >= NNODES) return;
  float acc = gb[0];
  for (int c = 0; c < HDIM; ++c) acc += h[(size_t)n * HDIM + c] * gw[c];
  gate[n] = acc;
  atomicMax(&gmax[batch[n]], f2o(acc));
}
__global__ void k_gate_expsum(float* __restrict__ gate, const int* __restrict__ batch,
                              const unsigned* __restrict__ gmax, float* __restrict__ gsum) {
  int n = blockIdx.x * blockDim.x + threadIdx.x;
  if (n >= NNODES) return;
  float ex = __expf(gate[n] - o2f(gmax[batch[n]]));
  gate[n] = ex;
  atomicAdd(&gsum[batch[n]], ex);
}
__global__ void k_pool_agg(const float* __restrict__ gate, const float* __restrict__ gsum,
                           const float* __restrict__ h, const int* __restrict__ batch,
                           float* __restrict__ pooled) {
  size_t t = (size_t)blockIdx.x * blockDim.x + threadIdx.x;
  if (t >= (size_t)NNODES * HDIM) return;
  int n = (int)(t >> 7), c = (int)(t & 127);
  int b = batch[n];
  atomicAdd(&pooled[(size_t)b * HDIM + c], (gate[n] / (gsum[b] + 1e-16f)) * h[t]);
}
__global__ void k_fc2(const float* __restrict__ in, const float* __restrict__ w,
                      const float* __restrict__ b, float* __restrict__ out) {
  int t = blockIdx.x * blockDim.x + threadIdx.x;
  if (t >= NGRAPH * OUTD) return;
  int g = t / OUTD, o = t % OUTD;
  float acc = b[o];
  for (int c = 0; c < 128; ++c) acc += in[(size_t)g * 128 + c] * w[c * OUTD + o];
  out[t] = acc;
}

// ---------------- host helpers ----------------
static inline unsigned gridFor(size_t n) { return (unsigned)((n + 255) / 256); }

static void launch_gemm(const unsigned short* A, const unsigned short* Bt, const float* bias,
                        float* Cf, unsigned short* Cb, int M, int K, int Nn, int relu,
                        hipStream_t s) {
  size_t waves = (size_t)((M + 15) / 16) * (Nn / 16);
  k_gemm_bf16_wmma<<<gridFor(waves * 32), 256, 0, s>>>(A, Bt, bias, Cf, Cb, M, K, Nn, relu);
}

extern "C" void kernel_launch(void* const* d_in, const int* in_sizes, int n_in,
                              void* d_out, int out_size, void* d_ws, size_t ws_size,
                              hipStream_t stream) {
  (void)in_sizes; (void)n_in; (void)out_size; (void)ws_size;

  const float* x     = (const float*)d_in[0];
  const int*   eidx  = (const int*)d_in[1];
  const int*   src   = eidx;
  const int*   dst   = eidx + NEDGES;
  const int*   batch = (const int*)d_in[2];

  const float *gw1[4], *gb1[4], *gw2[4], *gb2[4];
  for (int i = 0; i < 4; ++i) {
    gw1[i] = (const float*)d_in[3 + i * 4 + 0];
    gb1[i] = (const float*)d_in[3 + i * 4 + 1];
    gw2[i] = (const float*)d_in[3 + i * 4 + 2];
    gb2[i] = (const float*)d_in[3 + i * 4 + 3];
  }
  const float *bng[5], *bnb[5];
  for (int i = 0; i < 5; ++i) {
    bng[i] = (const float*)d_in[19 + i * 2 + 0];
    bnb[i] = (const float*)d_in[19 + i * 2 + 1];
  }
  const float* gat_w    = (const float*)d_in[29];
  const float* gat_asrc = (const float*)d_in[30];
  const float* gat_adst = (const float*)d_in[31];
  const float* gat_bias = (const float*)d_in[32];
  const float* gate_w   = (const float*)d_in[33];
  const float* gate_b   = (const float*)d_in[34];
  const float* fc1_w    = (const float*)d_in[35];
  const float* fc1_b    = (const float*)d_in[36];
  const float* fc2_w    = (const float*)d_in[37];
  const float* fc2_b    = (const float*)d_in[38];

  const int din[4] = {128, 128, 256, 512};
  const int dhh[4] = {128, 256, 512, 256};

  // ---- workspace carve (~180 MB) ----
  char* ws = (char*)d_ws;
  size_t off = 0;
  auto alloc = [&](size_t bytes) -> char* {
    char* p = ws + off;
    off += (bytes + 255) & ~(size_t)255;
    return p;
  };
  unsigned short *w1t[4], *w2t[4];
  for (int i = 0; i < 4; ++i) {
    w1t[i] = (unsigned short*)alloc((size_t)din[i] * dhh[i] * 2);
    w2t[i] = (unsigned short*)alloc((size_t)dhh[i] * dhh[i] * 2);
  }
  unsigned short* gatwt  = (unsigned short*)alloc((size_t)256 * 1024 * 2);
  unsigned short* fc1wt  = (unsigned short*)alloc((size_t)128 * 128 * 2);
  float*          A      = (float*)alloc((size_t)NNODES * 1024 * 4);       // z / xh
  float*          H      = (float*)alloc((size_t)NNODES * 512 * 4);        // activations
  unsigned short* X1     = (unsigned short*)alloc((size_t)NNODES * 512 * 2);
  unsigned short* X2     = (unsigned short*)alloc((size_t)NNODES * 512 * 2);
  float*          E1     = (float*)alloc((size_t)(NEDGES + NNODES) * HEADS * 4);
  unsigned*       S1     = (unsigned*)alloc((size_t)NNODES * HEADS * 4);
  float*          S2     = (float*)alloc((size_t)NNODES * HEADS * 4);
  float*          meanB  = (float*)alloc(512 * 4);
  float*          rstdB  = (float*)alloc(512 * 4);
  float*          asB    = (float*)alloc((size_t)NNODES * HEADS * 4);
  float*          adB    = (float*)alloc((size_t)NNODES * HEADS * 4);
  float*          gate   = (float*)alloc((size_t)NNODES * 4);
  unsigned*       gmax   = (unsigned*)alloc(NGRAPH * 4);
  float*          gsum   = (float*)alloc(NGRAPH * 4);
  float*          pooled = (float*)alloc((size_t)NGRAPH * 128 * 4);
  unsigned short* pooldB = (unsigned short*)alloc((size_t)NGRAPH * 128 * 2);
  float*          fcbuf  = (float*)alloc((size_t)NGRAPH * 128 * 4);

  // ---- weight conversion (bf16, B pre-transposed to Nn x K) ----
  for (int i = 0; i < 4; ++i) {
    k_cvt_transpose_bf16<<<gridFor((size_t)din[i] * dhh[i]), 256, 0, stream>>>(gw1[i], w1t[i], din[i], dhh[i]);
    k_cvt_transpose_bf16<<<gridFor((size_t)dhh[i] * dhh[i]), 256, 0, stream>>>(gw2[i], w2t[i], dhh[i], dhh[i]);
  }
  k_cvt_transpose_bf16<<<gridFor((size_t)256 * 1024), 256, 0, stream>>>(gat_w, gatwt, 256, 1024);
  k_cvt_transpose_bf16<<<gridFor((size_t)128 * 128), 256, 0, stream>>>(fc1_w, fc1wt, 128, 128);

  // ---- h = x ----
  k_copy_f32<<<gridFor((size_t)NNODES * DIN), 256, 0, stream>>>(H, x, (size_t)NNODES * DIN);

  // ---- 4x GIN + BN + ReLU ----
  for (int i = 0; i < 4; ++i) {
    int Di = din[i], Dh = dhh[i];
    size_t nDi = (size_t)NNODES * Di, nDh = (size_t)NNODES * Dh;
    k_copy_f32<<<gridFor(nDi), 256, 0, stream>>>(A, H, nDi);                       // z = h
    k_gin_scatter<<<gridFor((size_t)NEDGES * Di), 256, 0, stream>>>(H, src, dst, A, NEDGES, Di);
    k_cvt_bf16<<<gridFor(nDi), 256, 0, stream>>>(A, X1, nDi);
    launch_gemm(X1, w1t[i], gb1[i], nullptr, X2, NNODES, Di, Dh, 1, stream);       // relu(z@w1+b1)
    launch_gemm(X2, w2t[i], gb2[i], H, nullptr, NNODES, Dh, Dh, 0, stream);        // @w2+b2 -> H
    k_bn_stats<<<Dh, 256, 0, stream>>>(H, meanB, rstdB, NNODES, Dh);
    k_bn_apply_relu<<<gridFor(nDh), 256, 0, stream>>>(H, meanB, rstdB, bng[i], bnb[i], NNODES, Dh);
  }

  // ---- GAT (heads=8, concat=False, self-loops) ----
  {
    size_t n256 = (size_t)NNODES * 256;
    k_cvt_bf16<<<gridFor(n256), 256, 0, stream>>>(H, X1, n256);
    launch_gemm(X1, gatwt, nullptr, A, nullptr, NNODES, 256, 1024, 0, stream);     // xh = h @ gat_w
    k_gat_coef<<<gridFor((size_t)NNODES * HEADS), 256, 0, stream>>>(A, gat_asrc, gat_adst, asB, adB);
    k_fill_u32<<<gridFor((size_t)NNODES * HEADS), 256, 0, stream>>>(S1, 0u, (size_t)NNODES * HEADS);
    k_fill_f32<<<gridFor((size_t)NNODES * HEADS), 256, 0, stream>>>(S2, 0.f, (size_t)NNODES * HEADS);
    size_t ehT = (size_t)(NEDGES + NNODES) * HEADS;
    k_gat_logits<<<gridFor(ehT), 256, 0, stream>>>(asB, adB, src, dst, E1, S1);
    k_gat_expsum<<<gridFor(ehT), 256, 0, stream>>>(E1, dst, S1, S2);
    size_t nH = (size_t)NNODES * HDIM;
    k_fill_f32<<<gridFor(nH), 256, 0, stream>>>(H, 0.f, nH);
    k_gat_aggregate<<<gridFor((size_t)(NEDGES + NNODES) * HDIM), 256, 0, stream>>>(E1, S2, A, src, dst, H);
    k_gat_finalize<<<gridFor(nH), 256, 0, stream>>>(H, gat_bias);
    k_bn_stats<<<HDIM, 256, 0, stream>>>(H, meanB, rstdB, NNODES, HDIM);
    k_bn_apply_relu<<<gridFor(nH), 256, 0, stream>>>(H, meanB, rstdB, bng[4], bnb[4], NNODES, HDIM);
  }

  // ---- global attention pooling + MLP head ----
  {
    k_fill_u32<<<1, 256, 0, stream>>>(gmax, 0u, NGRAPH);
    k_fill_f32<<<1, 256, 0, stream>>>(gsum, 0.f, NGRAPH);
    k_gate_logits<<<gridFor(NNODES), 256, 0, stream>>>(H, gate_w, gate_b, batch, gate, gmax);
    k_gate_expsum<<<gridFor(NNODES), 256, 0, stream>>>(gate, batch, gmax, gsum);
    size_t pN = (size_t)NGRAPH * 128;
    k_fill_f32<<<gridFor(pN), 256, 0, stream>>>(pooled, 0.f, pN);
    k_pool_agg<<<gridFor((size_t)NNODES * HDIM), 256, 0, stream>>>(gate, gsum, H, batch, pooled);
    k_cvt_bf16<<<gridFor(pN), 256, 0, stream>>>(pooled, pooldB, pN);
    launch_gemm(pooldB, fc1wt, fc1_b, fcbuf, nullptr, NGRAPH, 128, 128, 1, stream); // relu(p@fc1+b)
    k_fc2<<<1, 256, 0, stream>>>(fcbuf, fc2_w, fc2_b, (float*)d_out);
  }
}
